// Grouped_Dilated_CSA_50371376447935
// MI455X (gfx1250) — matrix operand order repaired and checked by reading, not compiled, not run on hardware
//
#include <hip/hip_runtime.h>
#include <hip/hip_bf16.h>

// ---------------------------------------------------------------------------
// Grouped dilated channel-wise self-attention (XCA-style), CDNA5 / gfx1250.
// All GEMMs: v_wmma_f32_16x16x32_bf16, f32 accumulate.
// Fragment strategy: A operands pre-swizzled in global memory (32B/lane
// contiguous -> global_load_b128 pairs); B operands staged in LDS in
// fragment-major order (ds_store/ds_load b128); K4 reads fragments directly
// from global (K axis == contiguous pixel axis).
// ---------------------------------------------------------------------------

typedef __attribute__((ext_vector_type(16))) __bf16 v16bf;
typedef __attribute__((ext_vector_type(8)))  __bf16 v8bf;
typedef __attribute__((ext_vector_type(8)))  float  v8f;

#define BATCH 8
#define DIMC  192
#define C3    576          // 3*DIM
#define IMGH  128
#define IMGW  128
#define NPIX  (IMGH*IMGW)  // 16384
#define HEADS 8
#define CHH   24           // channels per head
#define DILN  2
#define KB_DIM (DIMC/32)   // 6 k-steps for K=192

__device__ __forceinline__ __bf16 f2bf(float f) { return (__bf16)f; }

// ---------------- K0: zero stats region ----------------
__global__ void k0_zero(float* __restrict__ p, int n) {
    int i = blockIdx.x * 256 + threadIdx.x;
    if (i < n) p[i] = 0.0f;
}

// ---------------- K1: swizzle w_qkv into fragment-major bf16 ----------------
// dst layout: [mt (C3/16)][kb (6)][lane (32)][e (16)]
__global__ void k1_swz(const float* __restrict__ src, __bf16* __restrict__ dst) {
    int i = blockIdx.x * 256 + threadIdx.x;
    if (i >= (C3 / 16) * KB_DIM * 512) return;
    int e  = i & 15;
    int l  = (i >> 4) & 31;
    int kb = (i >> 9) % KB_DIM;
    int mt = i / (512 * KB_DIM);
    int half = l >> 4, lm = l & 15;
    int k = kb * 32 + ((e < 8) ? (half * 8 + e) : (16 + half * 8 + (e - 8)));
    int m = mt * 16 + lm;
    dst[i] = f2bf(src[m * DIMC + k]);
}

// ---------------- K2: qkv GEMM  (C3 x 192) @ (192 x NPIX), per batch --------
// block 256 = 8 waves (2 M x 4 N), tile 32(M) x 64(N); grid (NPIX/64, C3/32, B)
__global__ __launch_bounds__(256)
void k2_qkv_gemm(const __bf16* __restrict__ WqSw,  // swizzled [C3/16][6][32][16]
                 const float*  __restrict__ x,     // [B][192][NPIX] f32
                 __bf16* __restrict__ qkv)         // [B][C3][NPIX] bf16
{
    __shared__ __align__(32) __bf16 Bf[4][32][16];   // fragment-major B tiles
    const int tid = threadIdx.x;
    const int n0  = blockIdx.x * 64;
    const int b   = blockIdx.z;

    const int wv   = tid >> 5;
    const int lane = tid & 31;
    const int half = lane >> 4;
    const int lm   = lane & 15;
    const int mt   = blockIdx.y * 2 + (wv >> 2);   // A fragment row-tile
    const int nt   = wv & 3;

    // staging role: column c (0..63), k-half, sub-8 group
    const int sc   = tid & 63;
    const int part = tid >> 6;         // 0..3
    const int shf  = part >> 1;        // k half (0/1)
    const int ssub = part & 1;         // element sub-group (0/1)

    v8f acc = {};
    for (int kb = 0; kb < KB_DIM; ++kb) {
        const int kk = kb * 32;
        // stage B fragment-major: one 16B LDS store per thread
        v8bf pk;
#pragma unroll
        for (int j = 0; j < 8; ++j)
            pk[j] = f2bf(x[((size_t)b * DIMC + kk + shf * 16 + ssub * 8 + j) * NPIX + n0 + sc]);
        *(v8bf*)&Bf[sc >> 4][ssub * 16 + (sc & 15)][shf * 8] = pk;
        __syncthreads();

        const v16bf a  = *(const v16bf*)&WqSw[(((size_t)mt * KB_DIM + kb) * 32 + lane) * 16];
        const v16bf bb = *(const v16bf*)&Bf[nt][lane][0];
        acc = __builtin_amdgcn_wmma_f32_16x16x32_bf16(false, a, false, bb,
                                                      (short)0, acc, false, false);
        __syncthreads();
    }
#pragma unroll
    for (int r = 0; r < 8; ++r) {
        int m = mt * 16 + half * 8 + r;
        int n = n0 + nt * 16 + lm;
        qkv[((size_t)b * C3 + m) * NPIX + n] = f2bf(acc[r]);
    }
}

// ---------------- K3: dilated depthwise 3x3 + q/k sumsq ----------------
__global__ __launch_bounds__(256)
void k3_dwconv(const __bf16* __restrict__ qkv,
               const float*  __restrict__ wdw,
               __bf16* __restrict__ qkv2,
               float* __restrict__ qn2, float* __restrict__ kn2)
{
    __shared__ float red[256];
    const int tid = threadIdx.x;
    const int c   = blockIdx.y;
    const int b   = blockIdx.z;
    const int pix = blockIdx.x * 256 + tid;
    const int py  = pix >> 7;
    const int px  = pix & 127;

    const __bf16* src = qkv + ((size_t)b * C3 + c) * NPIX;
    float w[9];
#pragma unroll
    for (int i = 0; i < 9; ++i) w[i] = wdw[c * 9 + i];

    float acc = 0.0f;
#pragma unroll
    for (int dy = -1; dy <= 1; ++dy)
#pragma unroll
        for (int dx = -1; dx <= 1; ++dx) {
            int yy = py + DILN * dy, xx = px + DILN * dx;
            if (yy >= 0 && yy < IMGH && xx >= 0 && xx < IMGW)
                acc += (float)src[yy * IMGW + xx] * w[(dy + 1) * 3 + (dx + 1)];
        }
    __bf16 v = f2bf(acc);
    qkv2[((size_t)b * C3 + c) * NPIX + pix] = v;

    if (c < 2 * DIMC) {
        float fv = (float)v;
        red[tid] = fv * fv;
        __syncthreads();
        for (int s = 128; s > 0; s >>= 1) {
            if (tid < s) red[tid] += red[tid + s];
            __syncthreads();
        }
        if (tid == 0) {
            float* dst = (c < DIMC) ? &qn2[b * DIMC + c] : &kn2[b * DIMC + (c - DIMC)];
            atomicAdd(dst, red[0]);
        }
    }
}

// ---------------- K4: Gram S = q . k^T per (b,head), fragments from global ---
// grid (NPIX/2048, B*HEADS), block 128 = 4 waves covering padded 32x32
__global__ __launch_bounds__(128)
void k4_gram(const __bf16* __restrict__ qkv2, float* __restrict__ gram)
{
    const int tid  = threadIdx.x;
    const int bh   = blockIdx.y;
    const int b    = bh >> 3;
    const int head = bh & 7;
    const int n0   = blockIdx.x * 2048;

    const __bf16* qbase = qkv2 + ((size_t)b * C3 + head * CHH) * NPIX;
    const __bf16* kbase = qkv2 + ((size_t)b * C3 + DIMC + head * CHH) * NPIX;

    const int wv   = tid >> 5;
    const int lane = tid & 31;
    const int half = lane >> 4;
    const int lm   = lane & 15;
    const int m0l  = (wv >> 1) * 16;
    const int n0l  = (wv & 1) * 16;

    const int rowc = m0l + lm;   // A row (q channel)
    const int rowd = n0l + lm;   // B col (k channel)
    const __bf16* qrow = qbase + (size_t)rowc * NPIX;
    const __bf16* krow = kbase + (size_t)rowd * NPIX;
    const bool qok = rowc < CHH, kok = rowd < CHH;

    v8f acc = {};
    for (int kk = 0; kk < 2048; kk += 32) {
        v16bf a = {};
        if (qok) {
            v8bf lo = *(const v8bf*)&qrow[n0 + kk + half * 16];
            v8bf hi = *(const v8bf*)&qrow[n0 + kk + half * 16 + 8];
            a = __builtin_shufflevector(lo, hi, 0, 1, 2, 3, 4, 5, 6, 7,
                                        8, 9, 10, 11, 12, 13, 14, 15);
        }
        v16bf bb = {};
        if (kok) bb = *(const v16bf*)&krow[n0 + kk + half * 16];
        acc = __builtin_amdgcn_wmma_f32_16x16x32_bf16(false, a, false, bb,
                                                      (short)0, acc, false, false);
    }
#pragma unroll
    for (int r = 0; r < 8; ++r) {
        int c = m0l + half * 8 + r;
        int d = n0l + lm;
        if (c < CHH && d < CHH)
            atomicAdd(&gram[(size_t)bh * (CHH * CHH) + c * CHH + d], acc[r]);
    }
}

// ---------------- K5: normalize + temperature + softmax ----------------
__global__ void k5_attn(const float* __restrict__ gram,
                        const float* __restrict__ qn2,
                        const float* __restrict__ kn2,
                        const float* __restrict__ temperature,
                        float* __restrict__ attn)
{
    const int bh   = blockIdx.x;
    const int b    = bh >> 3;
    const int head = bh & 7;
    const int c    = threadIdx.x;
    if (c >= CHH) return;
    const float temp = temperature[head];
    const float qn = fmaxf(sqrtf(qn2[b * DIMC + head * CHH + c]), 1e-12f);

    float a[CHH];
    float mx = -3.4e38f;
#pragma unroll
    for (int d = 0; d < CHH; ++d) {
        float kn = fmaxf(sqrtf(kn2[b * DIMC + head * CHH + d]), 1e-12f);
        a[d] = gram[(size_t)bh * (CHH * CHH) + c * CHH + d] / (qn * kn) * temp;
        mx = fmaxf(mx, a[d]);
    }
    float sum = 0.0f;
#pragma unroll
    for (int d = 0; d < CHH; ++d) { a[d] = __expf(a[d] - mx); sum += a[d]; }
    float inv = 1.0f / sum;
#pragma unroll
    for (int d = 0; d < CHH; ++d)
        attn[(size_t)bh * (CHH * CHH) + c * CHH + d] = a[d] * inv;
}

// ---------------- K6: W_b = Wproj @ blockdiag(attn_b), swizzled bf16 out -----
// grid B*DIMC (one output row), block 192 (one col hd each)
__global__ __launch_bounds__(192)
void k6_wb(const float* __restrict__ wproj,
           const float* __restrict__ attn,
           __bf16* __restrict__ WbSw)   // [B][DIMC/16][6][32][16] swizzled
{
    const int oc = blockIdx.x % DIMC;
    const int b  = blockIdx.x / DIMC;
    const int t  = threadIdx.x;          // hd = head*24 + d
    const int head = t / CHH;
    const int d    = t % CHH;
    float s = 0.0f;
#pragma unroll
    for (int ch = 0; ch < CHH; ++ch)
        s += wproj[oc * DIMC + head * CHH + ch] *
             attn[((size_t)(b * HEADS + head)) * (CHH * CHH) + ch * CHH + d];
    // scatter into fragment-major layout
    const int mt = oc >> 4, lm = oc & 15;
    const int kb = t >> 5,  kr = t & 31;
    int half, e;
    if      (kr < 8)  { half = 0; e = kr; }
    else if (kr < 16) { half = 1; e = kr - 8; }
    else if (kr < 24) { half = 0; e = 8 + (kr - 16); }
    else              { half = 1; e = 8 + (kr - 24); }
    const int lane = half * 16 + lm;
    WbSw[((((size_t)b * (DIMC / 16) + mt) * KB_DIM + kb) * 32 + lane) * 16 + e] = f2bf(s);
}

// ---------------- K7: final GEMM  out = W_b @ v, f32 out ----------------
// grid (NPIX/64, DIMC/32, B), block 256 (same tiling as K2)
__global__ __launch_bounds__(256)
void k7_out_gemm(const __bf16* __restrict__ WbSw,   // swizzled per batch
                 const __bf16* __restrict__ qkv2,   // v at channels 384..575
                 float* __restrict__ out)
{
    __shared__ __align__(32) __bf16 Bf[4][32][16];
    const int tid = threadIdx.x;
    const int n0  = blockIdx.x * 64;
    const int b   = blockIdx.z;

    const __bf16* A = WbSw + (size_t)b * (DIMC / 16) * KB_DIM * 512;
    const __bf16* V = qkv2 + ((size_t)b * C3 + 2 * DIMC) * NPIX;

    const int wv   = tid >> 5;
    const int lane = tid & 31;
    const int half = lane >> 4;
    const int lm   = lane & 15;
    const int mt   = blockIdx.y * 2 + (wv >> 2);
    const int nt   = wv & 3;

    const int sc   = tid & 63;
    const int part = tid >> 6;
    const int shf  = part >> 1;
    const int ssub = part & 1;

    v8f acc = {};
    for (int kb = 0; kb < KB_DIM; ++kb) {
        const int kk = kb * 32;
        v8bf pk;
#pragma unroll
        for (int j = 0; j < 8; ++j)
            pk[j] = V[(size_t)(kk + shf * 16 + ssub * 8 + j) * NPIX + n0 + sc];
        *(v8bf*)&Bf[sc >> 4][ssub * 16 + (sc & 15)][shf * 8] = pk;
        __syncthreads();

        const v16bf a  = *(const v16bf*)&A[(((size_t)mt * KB_DIM + kb) * 32 + lane) * 16];
        const v16bf bb = *(const v16bf*)&Bf[nt][lane][0];
        acc = __builtin_amdgcn_wmma_f32_16x16x32_bf16(false, a, false, bb,
                                                      (short)0, acc, false, false);
        __syncthreads();
    }
#pragma unroll
    for (int r = 0; r < 8; ++r) {
        int m = mt * 16 + half * 8 + r;
        int n = n0 + nt * 16 + lm;
        out[((size_t)b * DIMC + m) * NPIX + n] = acc[r];
    }
}

// ---------------------------------------------------------------------------
extern "C" void kernel_launch(void* const* d_in, const int* in_sizes, int n_in,
                              void* d_out, int out_size, void* d_ws, size_t ws_size,
                              hipStream_t stream) {
    const float* x           = (const float*)d_in[0];
    const float* w_qkv       = (const float*)d_in[1];
    const float* w_dw        = (const float*)d_in[2];
    const float* w_proj      = (const float*)d_in[3];
    const float* temperature = (const float*)d_in[4];
    float* out = (float*)d_out;

    char* base = (char*)d_ws;
    size_t o = 0;
    auto carve = [&](size_t bytes) -> char* {
        char* p = base + o;
        o = (o + bytes + 255) & ~(size_t)255;
        return p;
    };
    __bf16* qkv1    = (__bf16*)carve((size_t)BATCH * C3 * NPIX * 2);
    __bf16* qkv2    = (__bf16*)carve((size_t)BATCH * C3 * NPIX * 2);
    __bf16* wqkvSw  = (__bf16*)carve((size_t)(C3 / 16) * KB_DIM * 512 * 2);
    __bf16* WbSw    = (__bf16*)carve((size_t)BATCH * (DIMC / 16) * KB_DIM * 512 * 2);
    float*  stats   = (float*)carve((size_t)(BATCH * HEADS * CHH * CHH + 2 * BATCH * DIMC) * 4);
    float*  gram    = stats;
    float*  qn2     = stats + BATCH * HEADS * CHH * CHH;
    float*  kn2     = qn2 + BATCH * DIMC;
    float*  attn    = (float*)carve((size_t)BATCH * HEADS * CHH * CHH * 4);

    {   // K0: zero stats
        int n = BATCH * HEADS * CHH * CHH + 2 * BATCH * DIMC;
        k0_zero<<<(n + 255) / 256, 256, 0, stream>>>(stats, n);
    }
    {   // K1: swizzle qkv weights
        int n = (C3 / 16) * KB_DIM * 512;
        k1_swz<<<(n + 255) / 256, 256, 0, stream>>>(w_qkv, wqkvSw);
    }
    k2_qkv_gemm<<<dim3(NPIX / 64, C3 / 32, BATCH), 256, 0, stream>>>(wqkvSw, x, qkv1);
    k3_dwconv<<<dim3(NPIX / 256, C3, BATCH), 256, 0, stream>>>(qkv1, w_dw, qkv2, qn2, kn2);
    k4_gram<<<dim3(NPIX / 2048, BATCH * HEADS), 128, 0, stream>>>(qkv2, gram);
    k5_attn<<<BATCH * HEADS, 32, 0, stream>>>(gram, qn2, kn2, temperature, attn);
    k6_wb<<<BATCH * DIMC, DIMC, 0, stream>>>(w_proj, attn, WbSw);
    k7_out_gemm<<<dim3(NPIX / 64, DIMC / 32, BATCH), 256, 0, stream>>>(WbSw, qkv2, out);
}
